// Transformer_83116207112253
// MI455X (gfx1250) — matrix operand order repaired and checked by reading, not compiled, not run on hardware
//
#include <hip/hip_runtime.h>

typedef __attribute__((ext_vector_type(16))) _Float16 v16h;
typedef __attribute__((ext_vector_type(8)))  float    v8f;

#define DMODEL 512
#define NHEAD  8
#define DH     64

static __device__ __forceinline__ v8f wmma16(v16h a, v16h b, v8f c) {
  // D = A(16x32 f16) * B(32x16 f16) + C(16x16 f32)
  return __builtin_amdgcn_wmma_f32_16x16x32_f16(false, a, false, b, (short)0, c,
                                                false, false);
}

// Wave-local LDS ordering fence (LDS is in-order per wave; this also stops the
// compiler from reordering the ds accesses around it).
static __device__ __forceinline__ void wave_lds_fence() {
  asm volatile("s_wait_dscnt 0" ::: "memory");
}

// ds_swizzle_b32 xor-lane exchange: new_lane = ((lane & 0x1F) | 0) ^ xor_mask.
// PAT = (xor_mask << 10) | 0x1F.  One DS op, no per-step address math.
template <int PAT>
static __device__ __forceinline__ float ds_swz(float x) {
  return __int_as_float(__builtin_amdgcn_ds_swizzle(__float_as_int(x), PAT));
}
// Reduce over the 16 lanes sharing (lane>>4): xor over {1,2,4,8}.
static __device__ __forceinline__ float red16_max(float x) {
  x = fmaxf(x, ds_swz<0x041F>(x));
  x = fmaxf(x, ds_swz<0x081F>(x));
  x = fmaxf(x, ds_swz<0x101F>(x));
  x = fmaxf(x, ds_swz<0x201F>(x));
  return x;
}
static __device__ __forceinline__ float red16_sum(float x) {
  x += ds_swz<0x041F>(x);
  x += ds_swz<0x081F>(x);
  x += ds_swz<0x101F>(x);
  x += ds_swz<0x201F>(x);
  return x;
}

// ---------------------------------------------------------------------------
// Positional encoding: pe[pos, i] = sin(pos/10000^(i/256)) for i<256, cos(...)
// ---------------------------------------------------------------------------
__global__ void tk_pe(float* __restrict__ pe) {
  int pos = blockIdx.x;      // 0..511
  int i   = threadIdx.x;     // 0..255
  float ang = (float)pos * __powf(10000.f, -((float)i) / 256.f);
  pe[pos * DMODEL + i]       = __sinf(ang);
  pe[pos * DMODEL + 256 + i] = __cosf(ang);
}

// ---------------------------------------------------------------------------
// Embedding: out[row,:] = emb[tok[row],:]*sqrt(512) + PE[row % seq,:]
// ---------------------------------------------------------------------------
__global__ void tk_embed(const int* __restrict__ tok, const float* __restrict__ emb,
                         const float* __restrict__ pe, float* __restrict__ out,
                         int seq) {
  int row = blockIdx.x, tid = threadIdx.x;
  int t = tok[row];
  int pos = row % seq;
  const float sc = 22.62741699796952f;  // sqrt(512)
  size_t ro = (size_t)row * DMODEL;
  size_t eo = (size_t)t * DMODEL;
  size_t po = (size_t)pos * DMODEL;
  out[ro + tid]       = emb[eo + tid] * sc + pe[po + tid];
  out[ro + tid + 256] = emb[eo + tid + 256] * sc + pe[po + tid + 256];
}

// ---------------------------------------------------------------------------
// Residual + LayerNorm (eps = 1e-3): out = g*(v-mean)*rsqrt(var+eps)+b
// ---------------------------------------------------------------------------
__global__ void __launch_bounds__(256) tk_add_ln(
    const float* __restrict__ x, const float* __restrict__ res,
    const float* __restrict__ g, const float* __restrict__ b,
    float* __restrict__ out) {
  __shared__ float red[256];
  int row = blockIdx.x, tid = threadIdx.x;
  size_t ro = (size_t)row * DMODEL;
  float v0 = x[ro + tid] + res[ro + tid];
  float v1 = x[ro + tid + 256] + res[ro + tid + 256];
  red[tid] = v0 + v1;
  __syncthreads();
  for (int off = 128; off > 0; off >>= 1) {
    if (tid < off) red[tid] += red[tid + off];
    __syncthreads();
  }
  float mean = red[0] * (1.f / 512.f);
  __syncthreads();
  float d0 = v0 - mean, d1 = v1 - mean;
  red[tid] = d0 * d0 + d1 * d1;
  __syncthreads();
  for (int off = 128; off > 0; off >>= 1) {
    if (tid < off) red[tid] += red[tid + off];
    __syncthreads();
  }
  float rs = rsqrtf(red[0] * (1.f / 512.f) + 1e-3f);
  out[ro + tid]       = g[tid] * d0 * rs + b[tid];
  out[ro + tid + 256] = g[tid + 256] * d1 * rs + b[tid + 256];
}

// ---------------------------------------------------------------------------
// WMMA GEMM: C[M,N] = act(A[M,K] @ W[K,N] + bias[N]).  fp32 in/out, f16
// operands in LDS, f32 accumulation. Block tile 128x128, K-step 64; 8 waves
// in a 4x2 grid, each owns a 32x64 patch (2x4 WMMA tiles -> 8 wmma per
// 12 ds_load_b128, 16 wmma per LDS tile). NT=1 streams W non-temporally
// (single-use W_g is 256 MB > 192 MB L2).
// ---------------------------------------------------------------------------
#define GT_M 128
#define GT_N 128
#define GT_K 64
#define LK   72   // padded K stride (f16 elements)

template <int RELU, int NT>
__global__ void __launch_bounds__(256) tk_gemm(
    const float* __restrict__ A, const float* __restrict__ W,
    const float* __restrict__ bias, float* __restrict__ C,
    int M, int N, int K, int ldc) {
  __shared__ _Float16 sA[GT_M * LK];   // 18.4 KB
  __shared__ _Float16 sB[GT_N * LK];   // 18.4 KB, stored n-major: sB[n][k]
  const int tid  = threadIdx.x;
  const int lane = tid & 31;
  const int wid  = tid >> 5;
  const int m0 = blockIdx.y * GT_M;
  const int n0 = blockIdx.x * GT_N;
  const int wm = (wid >> 1) * 32;  // 0,32,64,96
  const int wn = (wid & 1) * 64;   // 0,64

  v8f acc[2][4];
#pragma unroll
  for (int i = 0; i < 2; i++)
#pragma unroll
    for (int j = 0; j < 4; j++) acc[i][j] = (v8f){};

  const int mlane = lane & 15;
  const int kb8   = (lane >> 4) * 8;
  const int kb16  = (lane >> 4) * 16;

  for (int k0 = 0; k0 < K; k0 += GT_K) {
    __syncthreads();
    {  // A tile: 128x64, 32 f16 per thread
      int row = tid >> 1, ks = (tid & 1) * 32;
      int gm = m0 + row;
      const float* ap = A + (size_t)gm * K + k0 + ks;
#pragma unroll
      for (int j = 0; j < 32; j++) {
        float v = (gm < M && (k0 + ks + j) < K) ? ap[j] : 0.f;
        sA[row * LK + ks + j] = (_Float16)v;
      }
    }
    {  // B tile: W[K,N] -> sB[n][k], 32 f16 per thread
      int nl = tid & 127;
      int kh = (tid >> 7) * 32;
      int gn = n0 + nl;
#pragma unroll
      for (int j = 0; j < 32; j++) {
        int gk = k0 + kh + j;
        float v = 0.f;
        if (gn < N && gk < K) {
          const float* wp = &W[(size_t)gk * N + gn];
          v = NT ? __builtin_nontemporal_load(wp) : *wp;
        }
        sB[nl * LK + kh + j] = (_Float16)v;
      }
    }
    if (k0 + GT_K < K) {  // prefetch next weight tile -> global_prefetch_b8
      __builtin_prefetch(W + (size_t)(k0 + GT_K) * N + n0 + (tid & 127), 0, 1);
    }
    __syncthreads();

#pragma unroll
    for (int kk = 0; kk < GT_K; kk += 32) {
      v16h aF[2], bF[4];
#pragma unroll
      for (int i = 0; i < 2; i++) {  // A frag: m=lane&15; half q: K=kb16+q
        const _Float16* ar = &sA[(wm + i * 16 + mlane) * LK + kk];
#pragma unroll
        for (int q = 0; q < 16; q++) {
          aF[i][q] = ar[kb16 + q];
        }
      }
#pragma unroll
      for (int j = 0; j < 4; j++) {  // B frag: n=lane&15; half q: K=kb16+q
        const _Float16* br = &sB[(wn + j * 16 + mlane) * LK + kk + kb16];
#pragma unroll
        for (int q = 0; q < 16; q++) bF[j][q] = br[q];
      }
#pragma unroll
      for (int i = 0; i < 2; i++)
#pragma unroll
        for (int j = 0; j < 4; j++) acc[i][j] = wmma16(aF[i], bF[j], acc[i][j]);
    }
  }

  const int rbase = (lane >> 4) * 8;  // C layout: m = rbase + r, n = lane&15
#pragma unroll
  for (int i = 0; i < 2; i++)
#pragma unroll
    for (int j = 0; j < 4; j++) {
      int gn = n0 + wn + j * 16 + mlane;
      if (gn >= N) continue;
      float bv = bias ? bias[gn] : 0.f;
#pragma unroll
      for (int r = 0; r < 8; r++) {
        int gm = m0 + wm + i * 16 + rbase + r;
        if (gm >= M) continue;
        float v = acc[i][j][r] + bv;
        if (RELU) v = v > 0.f ? v : 0.f;
        C[(size_t)gm * ldc + gn] = v;
      }
    }
}

// ---------------------------------------------------------------------------
// Flash attention. 256 threads = 8 waves per block; the block owns 128 query
// rows of one (batch, head); waves SHARE cooperatively loaded K/V chunks in
// LDS (8x less global K/V traffic). Each wave computes one 16-row q-tile:
// scores via WMMA (scale folded into Q), online softmax in the C-fragment
// layout (16-lane ds_swizzle xor reductions), P transposed through a
// per-wave LDS patch into the A layout, O accumulated via WMMA.
// kv batch = q batch >> kv_shift (cross-attn: 2 streams share one context).
// ---------------------------------------------------------------------------
#define AT_QT 128

__global__ void __launch_bounds__(256) tk_attn(
    const float* __restrict__ Q, const float* __restrict__ K,
    const float* __restrict__ V, float* __restrict__ O,
    int Tq, int Ts, int kv_shift, int causal) {
  __shared__ _Float16 sQ[AT_QT * 72];  // [q local][d]   18 KB
  __shared__ _Float16 sK[32 * 72];     // [key][d]       4.6 KB
  __shared__ _Float16 sVt[64 * 40];    // [d][key]       5.1 KB
  __shared__ _Float16 sP[8 * 16 * 32]; // per-wave P     8 KB

  const int tid   = threadIdx.x;
  const int lane  = tid & 31;
  const int wid   = tid >> 5;
  const int qbase = blockIdx.x * AT_QT;  // Tq is a multiple of 128
  const int h     = blockIdx.y;
  const int qb    = blockIdx.z;
  const int kb    = qb >> kv_shift;
  const int hof   = h * DH;
  const int mbase = qbase + wid * 16;    // this wave's q-tile

  {  // cooperative Q load (scaled by 1/sqrt(64)): 32 f16 per thread
    int m = tid >> 1, ds = (tid & 1) * 32;
    const float* qp = Q + ((size_t)(qb * Tq + qbase + m)) * DMODEL + hof + ds;
#pragma unroll
    for (int j = 0; j < 32; j++) sQ[m * 72 + ds + j] = (_Float16)(qp[j] * 0.125f);
  }
  __syncthreads();

  const int mlane = lane & 15;
  const int hi    = lane >> 4;
  const int kb8   = hi * 8;
  const int kb16  = hi * 16;

  v16h aQ[2];
#pragma unroll
  for (int i = 0; i < 2; i++) {
    const _Float16* qr = &sQ[(wid * 16 + mlane) * 72 + i * 32];
#pragma unroll
    for (int q = 0; q < 16; q++) {
      aQ[i][q] = qr[kb16 + q];
    }
  }

  float run_max[8], run_sum[8];
  v8f o[4];
#pragma unroll
  for (int r = 0; r < 8; r++) { run_max[r] = -1e30f; run_sum[r] = 0.f; }
#pragma unroll
  for (int t = 0; t < 4; t++) o[t] = (v8f){};

  int send_all = Ts;
  if (causal && (qbase + AT_QT) < send_all) send_all = qbase + AT_QT;
  int my_send = Ts;
  if (causal && (mbase + 16) < my_send) my_send = mbase + 16;
  _Float16* myP = &sP[wid * 512];

  for (int sbase = 0; sbase < send_all; sbase += 32) {
    __syncthreads();
    {  // cooperative K/V chunk load: 32 keys x 64 d, 8 f16 per thread each
      int key = tid >> 3, dseg = (tid & 7) * 8;
      int gs = sbase + key;
      bool ok = gs < Ts;
      const float* kp = K + ((size_t)(kb * Ts + gs)) * DMODEL + hof + dseg;
      const float* vp = V + ((size_t)(kb * Ts + gs)) * DMODEL + hof + dseg;
#pragma unroll
      for (int j = 0; j < 8; j++) {
        sK[key * 72 + dseg + j]    = ok ? (_Float16)kp[j] : (_Float16)0.f;
        sVt[(dseg + j) * 40 + key] = ok ? (_Float16)vp[j] : (_Float16)0.f;
      }
    }
    __syncthreads();

    if (sbase < my_send) {  // wave-uniform: safe around WMMA (EXEC all ones)
      // Scores: S(16x32 keys) = Q(16x64) * K^T
      v8f c0 = (v8f){}, c1 = (v8f){};
#pragma unroll
      for (int i = 0; i < 2; i++) {
        const _Float16* kr0 = &sK[mlane * 72 + i * 32 + kb16];
        const _Float16* kr1 = &sK[(16 + mlane) * 72 + i * 32 + kb16];
        v16h b0, b1;
#pragma unroll
        for (int q = 0; q < 16; q++) { b0[q] = kr0[q]; b1[q] = kr1[q]; }
        c0 = wmma16(aQ[i], b0, c0);
        c1 = wmma16(aQ[i], b1, c1);
      }

      // Online softmax in C layout (row m = 8*hi + r, col = lane&15)
      float pr0[8], pr1[8];
#pragma unroll
      for (int r = 0; r < 8; r++) {
        int qidx = mbase + hi * 8 + r;
        float s0 = c0[r], s1 = c1[r];
        int k0i = sbase + mlane, k1i = sbase + 16 + mlane;
        if (k0i >= Ts || (causal && k0i > qidx)) s0 = -1e9f;
        if (k1i >= Ts || (causal && k1i > qidx)) s1 = -1e9f;
        float mx = red16_max(fmaxf(s0, s1));
        float nm = fmaxf(run_max[r], mx);
        float alpha = __expf(run_max[r] - nm);
        run_max[r] = nm;
        float p0 = __expf(s0 - nm), p1 = __expf(s1 - nm);
        float ps = red16_sum(p0 + p1);
        run_sum[r] = run_sum[r] * alpha + ps;
#pragma unroll
        for (int t = 0; t < 4; t++) o[t][r] *= alpha;
        pr0[r] = p0;
        pr1[r] = p1;
      }

      // Transpose P through per-wave LDS patch into A-fragment layout
#pragma unroll
      for (int r = 0; r < 8; r++) {
        int m_loc = hi * 8 + r;
        myP[m_loc * 32 + mlane]      = (_Float16)pr0[r];
        myP[m_loc * 32 + 16 + mlane] = (_Float16)pr1[r];
      }
      wave_lds_fence();  // wave-private patch: LDS in-order per wave
      v16h aP;
      {
        const _Float16* pr = &myP[mlane * 32];
#pragma unroll
        for (int q = 0; q < 16; q++) {
          aP[q] = pr[kb16 + q];
        }
      }
      // O(16x64) += P(16x32) * V(32x64)
#pragma unroll
      for (int t = 0; t < 4; t++) {
        const _Float16* vr = &sVt[(t * 16 + mlane) * 40 + kb16];
        v16h bV;
#pragma unroll
        for (int q = 0; q < 16; q++) bV[q] = vr[q];
        o[t] = wmma16(aP, bV, o[t]);
      }
    }
  }

#pragma unroll
  for (int r = 0; r < 8; r++) {
    float inv = 1.f / run_sum[r];
    size_t gm = (size_t)(qb * Tq + mbase + hi * 8 + r) * DMODEL + hof;
#pragma unroll
    for (int t = 0; t < 4; t++) O[gm + t * 16 + mlane] = o[t][r] * inv;
  }
}

// ---------------------------------------------------------------------------
// Host orchestration
// ---------------------------------------------------------------------------
extern "C" void kernel_launch(void* const* d_in, const int* in_sizes, int n_in,
                              void* d_out, int out_size, void* d_ws, size_t ws_size,
                              hipStream_t stream) {
  (void)in_sizes; (void)n_in; (void)out_size; (void)ws_size;
  int idx = 0;
  auto FP = [&]() { return (const float*)d_in[idx++]; };
  const int* context = (const int*)d_in[idx++];
  const int* xtok    = (const int*)d_in[idx++];
  const float* enc_emb = FP();
  const float* dec_emb = FP();

  struct AttnP { const float *wq, *bq, *wk, *bk, *wv, *bv, *wo, *bo; };
  struct EncP { AttnP sa; const float *w1, *c1, *w2, *c2, *g1, *b1, *g2, *b2; };
  struct DecP {
    AttnP sa, ca;
    const float *w1, *c1, *w2, *c2, *g1, *b1, *g2, *b2, *g3, *b3;
  };
  auto readAttn = [&](AttnP& a) {
    a.wq = FP(); a.bq = FP(); a.wk = FP(); a.bk = FP();
    a.wv = FP(); a.bv = FP(); a.wo = FP(); a.bo = FP();
  };
  EncP enc[4];
  DecP dec[4];
  for (int l = 0; l < 4; l++) {
    readAttn(enc[l].sa);
    enc[l].w1 = FP(); enc[l].c1 = FP(); enc[l].w2 = FP(); enc[l].c2 = FP();
    enc[l].g1 = FP(); enc[l].b1 = FP(); enc[l].g2 = FP(); enc[l].b2 = FP();
  }
  for (int l = 0; l < 4; l++) {
    readAttn(dec[l].sa);
    readAttn(dec[l].ca);
    dec[l].w1 = FP(); dec[l].c1 = FP(); dec[l].w2 = FP(); dec[l].c2 = FP();
    dec[l].g1 = FP(); dec[l].b1 = FP(); dec[l].g2 = FP(); dec[l].b2 = FP();
    dec[l].g3 = FP(); dec[l].b3 = FP();
  }
  const float* W_o = FP();     const float* b_o = FP();
  const float* W_c = FP();     const float* b_c = FP();
  const float* W_g = FP();     const float* b_g = FP();
  const float* W_open = FP();  const float* b_open = FP();
  const float* W_close = FP(); const float* b_close = FP();

  // Workspace carve-out
  char* ws = (char*)d_ws;
  size_t off = 0;
  auto carve = [&](size_t bytes) {
    void* p = ws + off;
    off += (bytes + 255) & ~(size_t)255;
    return p;
  };
  float* pe   = (float*)carve((size_t)512 * 512 * 4);
  float* ctx  = (float*)carve((size_t)8192 * 512 * 4);
  float* qb   = (float*)carve((size_t)8192 * 512 * 4);
  float* kb   = (float*)carve((size_t)8192 * 512 * 4);
  float* vb   = (float*)carve((size_t)8192 * 512 * 4);
  float* t0   = (float*)carve((size_t)8192 * 512 * 4);
  float* t1   = (float*)carve((size_t)8192 * 2048 * 4);
  float* hbuf = (float*)carve((size_t)4096 * 512 * 4);
  float* flat = (float*)carve((size_t)16 * 32768 * 4);
  float* gbuf = (float*)carve((size_t)16 * 2048 * 4);

  auto gemm = [&](const float* A, const float* W, const float* bias, float* C,
                  int M, int N, int K, int ldc, bool relu, bool nt) {
    dim3 g((N + GT_N - 1) / GT_N, (M + GT_M - 1) / GT_M);
    if (relu)    tk_gemm<1, 0><<<g, 256, 0, stream>>>(A, W, bias, C, M, N, K, ldc);
    else if (nt) tk_gemm<0, 1><<<g, 256, 0, stream>>>(A, W, bias, C, M, N, K, ldc);
    else         tk_gemm<0, 0><<<g, 256, 0, stream>>>(A, W, bias, C, M, N, K, ldc);
  };
  auto attn = [&](const float* Qp, const float* Kp, const float* Vp, float* Op,
                  int Bq, int Tq, int Ts, int kvs, int causal) {
    dim3 g(Tq / AT_QT, NHEAD, Bq);
    tk_attn<<<g, 256, 0, stream>>>(Qp, Kp, Vp, Op, Tq, Ts, kvs, causal);
  };

  tk_pe<<<512, 256, 0, stream>>>(pe);

  // ---------------- Encoder ----------------
  const int ME = 16 * 512;
  tk_embed<<<ME, 256, 0, stream>>>(context, enc_emb, pe, ctx, 512);
  for (int l = 0; l < 4; l++) {
    gemm(ctx, enc[l].sa.wq, enc[l].sa.bq, qb, ME, 512, 512, 512, false, false);
    gemm(ctx, enc[l].sa.wk, enc[l].sa.bk, kb, ME, 512, 512, 512, false, false);
    gemm(ctx, enc[l].sa.wv, enc[l].sa.bv, vb, ME, 512, 512, 512, false, false);
    attn(qb, kb, vb, t1, 16, 512, 512, 0, 0);
    gemm(t1, enc[l].sa.wo, enc[l].sa.bo, t0, ME, 512, 512, 512, false, false);
    tk_add_ln<<<ME, 256, 0, stream>>>(t0, ctx, enc[l].g1, enc[l].b1, ctx);
    gemm(ctx, enc[l].w1, enc[l].c1, t1, ME, 2048, 512, 2048, true, false);
    gemm(t1, enc[l].w2, enc[l].c2, t0, ME, 512, 2048, 512, false, false);
    tk_add_ln<<<ME, 256, 0, stream>>>(t0, ctx, enc[l].g2, enc[l].b2, ctx);
  }

  // ---------------- Decoder (2 streams batched: bb = b*2 + stream) --------
  const int MD = 32 * 128;
  tk_embed<<<MD, 256, 0, stream>>>(xtok, dec_emb, pe, hbuf, 128);
  for (int l = 0; l < 4; l++) {
    // causal self-attention
    gemm(hbuf, dec[l].sa.wq, dec[l].sa.bq, qb, MD, 512, 512, 512, false, false);
    gemm(hbuf, dec[l].sa.wk, dec[l].sa.bk, kb, MD, 512, 512, 512, false, false);
    gemm(hbuf, dec[l].sa.wv, dec[l].sa.bv, vb, MD, 512, 512, 512, false, false);
    attn(qb, kb, vb, t1, 32, 128, 128, 0, 1);
    gemm(t1, dec[l].sa.wo, dec[l].sa.bo, t0, MD, 512, 512, 512, false, false);
    tk_add_ln<<<MD, 256, 0, stream>>>(t0, hbuf, dec[l].g1, dec[l].b1, hbuf);
    // cross-attention (kv batch = bb >> 1)
    gemm(hbuf, dec[l].ca.wq, dec[l].ca.bq, qb, MD, 512, 512, 512, false, false);
    gemm(ctx, dec[l].ca.wk, dec[l].ca.bk, kb, ME, 512, 512, 512, false, false);
    gemm(ctx, dec[l].ca.wv, dec[l].ca.bv, vb, ME, 512, 512, 512, false, false);
    attn(qb, kb, vb, t1, 32, 128, 512, 1, 0);
    gemm(t1, dec[l].ca.wo, dec[l].ca.bo, t0, MD, 512, 512, 512, false, false);
    tk_add_ln<<<MD, 256, 0, stream>>>(t0, hbuf, dec[l].g2, dec[l].b2, hbuf);
    // FFN
    gemm(hbuf, dec[l].w1, dec[l].c1, t1, MD, 2048, 512, 2048, true, false);
    gemm(t1, dec[l].w2, dec[l].c2, t0, MD, 512, 2048, 512, false, false);
    tk_add_ln<<<MD, 256, 0, stream>>>(t0, hbuf, dec[l].g3, dec[l].b3, hbuf);
  }

  // ---------------- Heads ----------------
  // flat[b, stream*T*VT + t*VT + v]  ==  rows ordered by (bb*T + t)
  for (int bb = 0; bb < 32; bb++) {
    const float* Wv = (bb & 1) ? W_c : W_o;
    const float* bv = (bb & 1) ? b_c : b_o;
    gemm(hbuf + (size_t)bb * 128 * 512, Wv, bv, flat + (size_t)bb * 128 * 128,
         128, 128, 512, 128, false, false);
  }
  // W_g is 256 MB fp32, read exactly once (> 192 MB L2): stream non-temporally.
  gemm(flat, W_g, b_g, gbuf, 16, 2048, 32768, 2048, false, true);
  float* out = (float*)d_out;  // [B, 2, VT]
  gemm(gbuf, W_open, b_open, out, 16, 128, 2048, 256, false, false);
  gemm(gbuf, W_close, b_close, out + 128, 16, 128, 2048, 256, false, false);
}
